// MDNV2_39067022524810
// MI455X (gfx1250) — matrix operand order repaired and checked by reading, not compiled, not run on hardware
//
#include <hip/hip_runtime.h>
#include <math.h>

typedef __attribute__((ext_vector_type(16))) _Float16 v16h;
typedef __attribute__((ext_vector_type(8)))  _Float16 v8h;
typedef __attribute__((ext_vector_type(8)))  float    v8f;

#define CC   128      // C
#define HH   256      // H
#define GA   140      // G*A
#define AA   14       // A
#define NCOL 448      // 420 padded to 28*16 (7 quad-groups of col-tiles)
#define MT   64       // pair rows per workgroup
#define LDA  264      // LDS A row stride in halves (256 + 8 pad)
#define BNL  1024     // B*NL
#define BNP  8192     // B*NP

__device__ __forceinline__ float elu1(float x) {
    return x > 0.f ? x : (expf(x) - 1.f);
}

// ---------- tiny prep kernels ----------

__global__ void k_invmap(const int* __restrict__ idx, int* __restrict__ inv, int n) {
    int i = blockIdx.x * blockDim.x + threadIdx.x;
    if (i < n) inv[idx[i]] = i;
}

__global__ void k_affine(const float* __restrict__ g, const float* __restrict__ be,
                         const float* __restrict__ rm, const float* __restrict__ rv,
                         const float* __restrict__ b1,
                         float* __restrict__ alpha, float* __restrict__ cshift) {
    int h = threadIdx.x;
    if (h < HH) {
        float a = g[h] * rsqrtf(rv[h] + 1e-5f);
        alpha[h]  = a;
        cshift[h] = a * (b1[h] - rm[h]) + be[h];
    }
}

// Build transposed, f16, fused head weight matrix WT[n][k] (n: output col, k contiguous)
__global__ void k_prep_whead(const float* __restrict__ Wpi, const float* __restrict__ bpi,
                             const float* __restrict__ Wsig, const float* __restrict__ bsig,
                             const float* __restrict__ Wmu, const float* __restrict__ bmu,
                             _Float16* __restrict__ WT, float* __restrict__ bh) {
    int id = blockIdx.x * blockDim.x + threadIdx.x; // n*HH + k
    if (id >= NCOL * HH) return;
    int n = id / HH, k = id % HH;
    float w = 0.f, b = 0.f;
    if (n < GA)          { w = Wpi [k * GA + n];          b = bpi [n]; }
    else if (n < 2 * GA) { w = Wsig[k * GA + (n - GA)];   b = bsig[n - GA]; }
    else if (n < 3 * GA) { w = Wmu [k * GA + (n - 2*GA)]; b = bmu [n - 2*GA]; }
    WT[n * HH + k] = (_Float16)w;
    if (k == 0) bh[n] = b;
}

// xl = h_l @ W1[0:128,:] ; xp = h_p @ W1[128:256,:]  (tiny GEMMs)
__global__ void k_feat_gemm(const float* __restrict__ F, const float* __restrict__ W1,
                            float* __restrict__ out, int rows, int koff) {
    int id = blockIdx.x * blockDim.x + threadIdx.x;
    if (id >= rows * HH) return;
    int r = id / HH, h = id % HH;
    const float* f = F + r * CC;
    float acc = 0.f;
    #pragma unroll 8
    for (int k = 0; k < CC; ++k)
        acc += f[k] * W1[(koff + k) * HH + h];
    out[r * HH + h] = acc;
}

// ---------- main fused head GEMM (WMMA) ----------

__device__ __forceinline__ void head_epilogue(
    const v8f& acc, int cg, int rbase, int M, const float* __restrict__ bh,
    float* __restrict__ out_pi, float* __restrict__ out_sig, float* __restrict__ out_mu) {
    if (cg < 3 * GA) {
        float bias = bh[cg];
        #pragma unroll
        for (int i = 0; i < 8; ++i) {
            int rr = rbase + i;
            if (rr < M) {
                float v = acc[i] + bias;
                if (cg < GA)          out_pi [rr * GA + cg]            = v;             // logits
                else if (cg < 2 * GA) out_sig[rr * GA + (cg - GA)]     = elu1(v) + 1.1f;
                else                  out_mu [rr * GA + (cg - 2 * GA)] = elu1(v) + 1.0f;
            }
        }
    }
}

__global__ __launch_bounds__(256) void k_head(
    const float* __restrict__ xl, const float* __restrict__ xp,
    const int* __restrict__ inv_l, const int* __restrict__ inv_p,
    const int* __restrict__ pair_l, const int* __restrict__ pair_p,
    const float* __restrict__ alpha, const float* __restrict__ cshift,
    const _Float16* __restrict__ WT, const float* __restrict__ bh,
    float* __restrict__ out_pi, float* __restrict__ out_sig, float* __restrict__ out_mu,
    int M)
{
    __shared__ _Float16 sA[MT * LDA];
    const int tid = threadIdx.x;

    // Phase 1: build f16 activation tile (affine-BN + ELU of xl+xp gather), 16B packed stores
    {
        int r = tid >> 2, q = tid & 3;            // 64 rows x 4 quarter-columns (64 each)
        int m = blockIdx.x * MT + r;
        if (m < M) {
            int li = inv_l[pair_l[m]];
            int pidx = inv_p[pair_p[m]];
            const float* xlr = xl + (size_t)li * HH + q * 64;
            const float* xpr = xp + (size_t)pidx * HH + q * 64;
            const float* al  = alpha  + q * 64;
            const float* cs  = cshift + q * 64;
            #pragma unroll
            for (int h8 = 0; h8 < 8; ++h8) {
                v8h pack;
                #pragma unroll
                for (int j = 0; j < 8; ++j) {
                    int hh = h8 * 8 + j;
                    float v = al[hh] * (xlr[hh] + xpr[hh]) + cs[hh];
                    pack[j] = (_Float16)elu1(v);
                }
                *(v8h*)&sA[r * LDA + q * 64 + h8 * 8] = pack;
            }
        } else {
            v8h z = {};
            #pragma unroll
            for (int h8 = 0; h8 < 8; ++h8)
                *(v8h*)&sA[r * LDA + q * 64 + h8 * 8] = z;
        }
    }
    __syncthreads();

    // Phase 2: 8 waves sweep 4 row-tiles x 7 col-quads; each unit = 16x64 output,
    // K=256 in 8 k-steps, 4 WMMAs per A-fragment load (A reused across 4 col-tiles).
    const int wave = tid >> 5, lane = tid & 31;
    const int lrow = lane & 15, grp = lane >> 4;
    for (int t = wave; t < (MT / 16) * (NCOL / 64); t += 8) {   // 4*7 = 28 units
        int rt = t / (NCOL / 64), cq = t % (NCOL / 64);
        v8f acc0 = {}, acc1 = {}, acc2 = {}, acc3 = {};
        const int arow = rt * 16 + lrow;
        const _Float16* w0 = WT + ((size_t)(cq * 64 +  0) + lrow) * HH;
        const _Float16* w1 = WT + ((size_t)(cq * 64 + 16) + lrow) * HH;
        const _Float16* w2 = WT + ((size_t)(cq * 64 + 32) + lrow) * HH;
        const _Float16* w3 = WT + ((size_t)(cq * 64 + 48) + lrow) * HH;
        #pragma unroll
        for (int kk = 0; kk < 8; ++kk) {
            int kb = kk * 32;
            // A fragment: lanes 0-15 -> K {kb+0..15}; lanes 16-31 -> K {kb+16..31}
            v8h lo = *(const v8h*)&sA[arow * LDA + kb + 16 * grp];
            v8h hi = *(const v8h*)&sA[arow * LDA + kb + 16 * grp + 8];
            v16h a = __builtin_shufflevector(lo, hi, 0,1,2,3,4,5,6,7,8,9,10,11,12,13,14,15);
            // B fragments: column n = base + lrow, 16 contiguous K per lane group
            v16h b0 = *(const v16h*)(w0 + kb + 16 * grp);
            v16h b1 = *(const v16h*)(w1 + kb + 16 * grp);
            v16h b2 = *(const v16h*)(w2 + kb + 16 * grp);
            v16h b3 = *(const v16h*)(w3 + kb + 16 * grp);
            acc0 = __builtin_amdgcn_wmma_f32_16x16x32_f16(false, a, false, b0, (short)0, acc0, false, false);
            acc1 = __builtin_amdgcn_wmma_f32_16x16x32_f16(false, a, false, b1, (short)0, acc1, false, false);
            acc2 = __builtin_amdgcn_wmma_f32_16x16x32_f16(false, a, false, b2, (short)0, acc2, false, false);
            acc3 = __builtin_amdgcn_wmma_f32_16x16x32_f16(false, a, false, b3, (short)0, acc3, false, false);
        }
        // Epilogue: C/D layout -> lane holds col (lane&15), rows i + 8*(lane>>4)
        int rbase = blockIdx.x * MT + rt * 16 + grp * 8;
        head_epilogue(acc0, cq * 64 +  0 + lrow, rbase, M, bh, out_pi, out_sig, out_mu);
        head_epilogue(acc1, cq * 64 + 16 + lrow, rbase, M, bh, out_pi, out_sig, out_mu);
        head_epilogue(acc2, cq * 64 + 32 + lrow, rbase, M, bh, out_pi, out_sig, out_mu);
        head_epilogue(acc3, cq * 64 + 48 + lrow, rbase, M, bh, out_pi, out_sig, out_mu);
    }
}

// In-place softmax over 140 pi logits, one wave32 per row
__global__ __launch_bounds__(256) void k_softmax(float* __restrict__ pi, int M) {
    int wave = threadIdx.x >> 5, lane = threadIdx.x & 31;
    int row = blockIdx.x * 8 + wave;
    if (row >= M) return;
    float* p = pi + (size_t)row * GA;
    float mx = -1e30f;
    for (int c = lane; c < GA; c += 32) mx = fmaxf(mx, p[c]);
    for (int o = 16; o > 0; o >>= 1) mx = fmaxf(mx, __shfl_xor(mx, o, 32));
    float s = 0.f;
    for (int c = lane; c < GA; c += 32) s += expf(p[c] - mx);
    for (int o = 16; o > 0; o >>= 1) s += __shfl_xor(s, o, 32);
    float inv = 1.f / s;
    for (int c = lane; c < GA; c += 32) p[c] = expf(p[c] - mx) * inv;
}

// ---------- side outputs ----------

__global__ void k_dist(const float* __restrict__ lig, const float* __restrict__ prot,
                       const int* __restrict__ inv_l, const int* __restrict__ inv_p,
                       const int* __restrict__ pair_l, const int* __restrict__ pair_p,
                       float* __restrict__ out, int M) {
    long id = blockIdx.x * (long)blockDim.x + threadIdx.x;
    if (id >= (long)M * AA) return;
    int m = (int)(id / AA), a = (int)(id % AA);
    int li = inv_l[pair_l[m]], pidx = inv_p[pair_p[m]];
    const float* X = lig + (size_t)li * 3;
    const float* Y = prot + ((size_t)pidx * AA + a) * 3;
    double dx = (double)X[0] - (double)Y[0];
    double dy = (double)X[1] - (double)Y[1];
    double dz = (double)X[2] - (double)Y[2];
    out[(long)m * AA + a] = (float)sqrt(dx * dx + dy * dy + dz * dz);
}

__global__ void k_atom(const float* __restrict__ h_l, const float* __restrict__ Wat,
                       const float* __restrict__ bat, float* __restrict__ out, int TL) {
    int id = blockIdx.x * blockDim.x + threadIdx.x;
    if (id >= TL * 17) return;
    int r = id / 17, c = id % 17;
    const float* f = h_l + (size_t)r * CC;
    float acc = bat[c];
    #pragma unroll 8
    for (int k = 0; k < CC; ++k) acc += f[k] * Wat[k * 17 + c];
    out[id] = acc;
}

__global__ void k_bond(const float* __restrict__ h_l, const int* __restrict__ src,
                       const int* __restrict__ dst, const float* __restrict__ Wbt,
                       const float* __restrict__ bbt, float* __restrict__ out, int E) {
    int id = blockIdx.x * blockDim.x + threadIdx.x;
    if (id >= E * 5) return;
    int e = id / 5, c = id % 5;
    const float* fs = h_l + (size_t)src[e] * CC;
    const float* fd = h_l + (size_t)dst[e] * CC;
    float acc = bbt[c];
    #pragma unroll 8
    for (int k = 0; k < CC; ++k)
        acc += fs[k] * Wbt[k * 5 + c] + fd[k] * Wbt[(CC + k) * 5 + c];
    out[id] = acc;
}

__global__ void k_pairb(const int* __restrict__ pb, float* __restrict__ out, int M) {
    int i = blockIdx.x * blockDim.x + threadIdx.x;
    if (i < M) out[i] = (float)pb[i];
}

// ---------- launcher ----------

extern "C" void kernel_launch(void* const* d_in, const int* in_sizes, int n_in,
                              void* d_out, int out_size, void* d_ws, size_t ws_size,
                              hipStream_t stream) {
    (void)n_in; (void)out_size; (void)ws_size;
    const float* h_l      = (const float*)d_in[0];
    const float* h_p      = (const float*)d_in[1];
    const float* lig_pos  = (const float*)d_in[2];
    const float* prot_pos = (const float*)d_in[3];
    const int*   edge_src = (const int*)d_in[4];
    const int*   edge_dst = (const int*)d_in[5];
    const int*   idx_l    = (const int*)d_in[6];
    const int*   idx_p    = (const int*)d_in[7];
    const int*   pair_l   = (const int*)d_in[9];
    const int*   pair_p   = (const int*)d_in[10];
    const int*   pair_b   = (const int*)d_in[11];
    const float* W1   = (const float*)d_in[12];
    const float* b1   = (const float*)d_in[13];
    const float* gam  = (const float*)d_in[14];
    const float* bet  = (const float*)d_in[15];
    const float* rm   = (const float*)d_in[16];
    const float* rv   = (const float*)d_in[17];
    const float* Wpi  = (const float*)d_in[18];
    const float* bpi  = (const float*)d_in[19];
    const float* Wsig = (const float*)d_in[20];
    const float* bsig = (const float*)d_in[21];
    const float* Wmu  = (const float*)d_in[22];
    const float* bmu  = (const float*)d_in[23];
    const float* Wat  = (const float*)d_in[24];
    const float* bat  = (const float*)d_in[25];
    const float* Wbt  = (const float*)d_in[26];
    const float* bbt  = (const float*)d_in[27];

    const int TL = in_sizes[0] / CC;
    const int TP = in_sizes[1] / CC;
    const int M  = in_sizes[9];
    const int E  = in_sizes[4];

    // workspace carve-out (256B aligned slices)
    char* w = (char*)d_ws;
    size_t off = 0;
    auto carve = [&](size_t bytes) -> void* {
        off = (off + 255) & ~(size_t)255;
        void* p = w + off;
        off += bytes;
        return p;
    };
    int*       inv_l  = (int*)carve((size_t)BNL * 4);
    int*       inv_p  = (int*)carve((size_t)BNP * 4);
    float*     alpha  = (float*)carve(HH * 4);
    float*     cshift = (float*)carve(HH * 4);
    float*     bh     = (float*)carve(NCOL * 4);
    _Float16*  WT     = (_Float16*)carve((size_t)NCOL * HH * 2);
    float*     xl     = (float*)carve((size_t)TL * HH * 4);
    float*     xp     = (float*)carve((size_t)TP * HH * 4);

    // output carve-out (flat return order)
    float* out = (float*)d_out;
    float* out_pi   = out;
    float* out_sig  = out + (size_t)M * GA;
    float* out_mu   = out + (size_t)2 * M * GA;
    float* out_dist = out + (size_t)3 * M * GA;
    float* out_at   = out_dist + (size_t)M * AA;
    float* out_bt   = out_at + (size_t)TL * 17;
    float* out_pb   = out_bt + (size_t)E * 5;

    const int T = 256;
    k_invmap<<<(TL + T - 1) / T, T, 0, stream>>>(idx_l, inv_l, TL);
    k_invmap<<<(TP + T - 1) / T, T, 0, stream>>>(idx_p, inv_p, TP);
    k_affine<<<1, T, 0, stream>>>(gam, bet, rm, rv, b1, alpha, cshift);
    k_prep_whead<<<(NCOL * HH + T - 1) / T, T, 0, stream>>>(Wpi, bpi, Wsig, bsig, Wmu, bmu, WT, bh);
    k_feat_gemm<<<((size_t)TL * HH + T - 1) / T, T, 0, stream>>>(h_l, W1, xl, TL, 0);
    k_feat_gemm<<<((size_t)TP * HH + T - 1) / T, T, 0, stream>>>(h_p, W1, xp, TP, CC);

    k_head<<<(M + MT - 1) / MT, T, 0, stream>>>(xl, xp, inv_l, inv_p, pair_l, pair_p,
                                                alpha, cshift, WT, bh,
                                                out_pi, out_sig, out_mu, M);
    k_softmax<<<(M + 7) / 8, T, 0, stream>>>(out_pi, M);

    k_dist<<<(int)(((long)M * AA + T - 1) / T), T, 0, stream>>>(
        lig_pos, prot_pos, inv_l, inv_p, pair_l, pair_p, out_dist, M);
    k_atom<<<(TL * 17 + T - 1) / T, T, 0, stream>>>(h_l, Wat, bat, out_at, TL);
    k_bond<<<(E * 5 + T - 1) / T, T, 0, stream>>>(h_l, edge_src, edge_dst, Wbt, bbt, out_bt, E);
    k_pairb<<<(M + T - 1) / T, T, 0, stream>>>(pair_b, out_pb, M);
}
